// _MultiHeadSelfAttention_55628416418161
// MI455X (gfx1250) — matrix operand order repaired and checked by reading, not compiled, not run on hardware
//
#include <hip/hip_runtime.h>

typedef __attribute__((ext_vector_type(16))) __bf16 bf16x16;
typedef __attribute__((ext_vector_type(8)))  __bf16 bf16x8;
typedef __attribute__((ext_vector_type(2)))  __bf16 bf16x2;
typedef __attribute__((ext_vector_type(8)))  float  f32x8;
typedef __attribute__((ext_vector_type(4)))  float  f32x4;

#define DMODEL 1024
#define NHEADS 16
#define HDIM   64
#define BATCH  4
#define SEQ    2048
#define ROWS   (BATCH*SEQ)   // 8192

__device__ __forceinline__ f32x8 wmma_bf16(bf16x16 a, bf16x16 b, f32x8 c) {
  // D = A(16x32 bf16) * B(32x16 bf16) + C(16x16 f32)
  return __builtin_amdgcn_wmma_f32_16x16x32_bf16(false, a, false, b, (short)0, c, false, false);
}

__device__ __forceinline__ bf16x16 join8(bf16x8 lo, bf16x8 hi) {
  bf16x16 r;
#pragma unroll
  for (int i = 0; i < 8; ++i) { r[i] = lo[i]; r[i + 8] = hi[i]; }
  return r;
}

// gfx1250 async global->LDS copy (ASYNCcnt path), 16 bytes per lane.
__device__ __forceinline__ void async_copy_b128(const void* gptr, void* lptr) {
  unsigned lds = (unsigned)(unsigned long long)lptr;  // low 32b of generic = LDS offset
  asm volatile("global_load_async_to_lds_b128 %0, %1, off"
               :: "v"(lds), "v"(gptr) : "memory");
}
__device__ __forceinline__ void wait_async0() {
  asm volatile("s_wait_asynccnt 0" ::: "memory");
}

// ---------------------------------------------------------------------------
// Kernel 1: fused QKV projection.  C[8192,3072] = X[8192,1024] @ Wqkv + bias.
// 128x128 block tile, K staged 64 at a time, software-pipelined global loads.
// 8 waves (4x2); wave = 2x4 tiles of 16x16; 16 WMMA per stage.
// W is transposed into LDS with packed bf16x2 (b32) stores (2 k-rows/thread).
// Output scattered to Q/K/V bf16 tensors [B,H,T,64]; Q pre-scaled by 1/8.
// ---------------------------------------------------------------------------
__global__ __launch_bounds__(256)
void qkv_gemm_kernel(const float* __restrict__ x, const float* __restrict__ w,
                     const float* __restrict__ bias,
                     __bf16* __restrict__ Q, __bf16* __restrict__ K,
                     __bf16* __restrict__ V)
{
  constexpr int KD  = DMODEL;      // 1024
  constexpr int N   = 3 * DMODEL;  // 3072
  constexpr int LDT = 72;          // 64 + 8 pad
  __shared__ __bf16 lA[128 * LDT]; // [m][k]
  __shared__ __bf16 lB[128 * LDT]; // [n][k] (W transposed)

  const int tid  = threadIdx.x;
  const int lane = tid & 31;
  const int wave = tid >> 5;
  const int wm   = wave >> 1;      // 0..3
  const int wn   = wave & 1;       // 0..1
  const int lo16 = lane & 15;
  const int hi   = lane >> 4;

  const int m0 = blockIdx.y * 128;
  const int n0 = blockIdx.x * 128;

  f32x8 acc[2][4];
#pragma unroll
  for (int i = 0; i < 2; ++i)
#pragma unroll
    for (int j = 0; j < 4; ++j) acc[i][j] = {};

  const int arow = tid >> 1;          // 0..127
  const int acol = (tid & 1) * 32;    // 0 / 32
  const int wkp  = tid >> 3;          // k-pair 0..31 -> rows {2wkp, 2wkp+1}
  const int ncol = (tid & 7) * 16;    // 0..112

  // prologue: preload first K-slab into registers
  f32x4 pa[8], pw0[4], pw1[4];
  {
    const float* asrc = x + (size_t)(m0 + arow) * KD + acol;
    const float* wsrc = w + (size_t)(2 * wkp) * N + n0 + ncol;
#pragma unroll
    for (int i = 0; i < 8; ++i) pa[i] = *(const f32x4*)(asrc + 4 * i);
#pragma unroll
    for (int i = 0; i < 4; ++i) pw0[i] = *(const f32x4*)(wsrc + 4 * i);
#pragma unroll
    for (int i = 0; i < 4; ++i) pw1[i] = *(const f32x4*)(wsrc + N + 4 * i);
  }

  for (int kk = 0; kk < KD; kk += 64) {
    // commit prefetched slab to LDS (fp32 -> bf16)
#pragma unroll
    for (int j = 0; j < 4; ++j) {
      bf16x8 t;
#pragma unroll
      for (int i = 0; i < 8; ++i) t[i] = (__bf16)pa[2 * j + (i >> 2)][i & 3];
      *(bf16x8*)&lA[arow * LDT + acol + 8 * j] = t;
    }
#pragma unroll
    for (int i = 0; i < 16; ++i) {   // packed pair store along k (b32)
      bf16x2 t;
      t[0] = (__bf16)pw0[i >> 2][i & 3];
      t[1] = (__bf16)pw1[i >> 2][i & 3];
      *(bf16x2*)&lB[(ncol + i) * LDT + 2 * wkp] = t;
    }
    __syncthreads();

    // prefetch next slab while this one is consumed by WMMAs
    if (kk + 64 < KD) {
      const float* asrc = x + (size_t)(m0 + arow) * KD + kk + 64 + acol;
      const float* wsrc = w + (size_t)(kk + 64 + 2 * wkp) * N + n0 + ncol;
#pragma unroll
      for (int i = 0; i < 8; ++i) pa[i] = *(const f32x4*)(asrc + 4 * i);
#pragma unroll
      for (int i = 0; i < 4; ++i) pw0[i] = *(const f32x4*)(wsrc + 4 * i);
#pragma unroll
      for (int i = 0; i < 4; ++i) pw1[i] = *(const f32x4*)(wsrc + N + 4 * i);
    }

#pragma unroll
    for (int ks = 0; ks < 2; ++ks) {
      bf16x16 afrag[2], bfrag[4];
#pragma unroll
      for (int mt = 0; mt < 2; ++mt) {           // A layout (ISA 7.12.2)
        const int row = wm * 32 + mt * 16 + lo16;
        bf16x8 g0 = *(const bf16x8*)&lA[row * LDT + ks * 32 + 8 * hi];
        bf16x8 g1 = *(const bf16x8*)&lA[row * LDT + ks * 32 + 16 + 8 * hi];
        afrag[mt] = join8(g0, g1);
      }
#pragma unroll
      for (int nt = 0; nt < 4; ++nt) {           // B layout: same k order as A
        const int col = wn * 64 + nt * 16 + lo16;
        bf16x8 g0 = *(const bf16x8*)&lB[col * LDT + ks * 32 + 8 * hi];
        bf16x8 g1 = *(const bf16x8*)&lB[col * LDT + ks * 32 + 16 + 8 * hi];
        bfrag[nt] = join8(g0, g1);
      }
#pragma unroll
      for (int mt = 0; mt < 2; ++mt)
#pragma unroll
        for (int nt = 0; nt < 4; ++nt)
          acc[mt][nt] = wmma_bf16(afrag[mt], bfrag[nt], acc[mt][nt]);
    }
    __syncthreads();
  }

  // epilogue: bias, scale Q, scatter to [B,H,T,64] bf16
#pragma unroll
  for (int nt = 0; nt < 4; ++nt) {
    const int n = n0 + wn * 64 + nt * 16 + lo16;
    const float bv = bias[n];
    const int which = n >> 10;                  // 0=Q 1=K 2=V (uniform per tile)
    const int h = (n >> 6) & (NHEADS - 1);
    const int d = n & (HDIM - 1);
    __bf16* dst = (which == 0) ? Q : (which == 1) ? K : V;
    const float sc = (which == 0) ? 0.125f : 1.0f;   // 1/sqrt(64) folded into Q
#pragma unroll
    for (int mt = 0; mt < 2; ++mt)
#pragma unroll
      for (int r = 0; r < 8; ++r) {
        const int m  = m0 + wm * 32 + mt * 16 + r + 8 * hi;
        const int bb = m >> 11;
        const int t  = m & (SEQ - 1);
        dst[(((size_t)(bb * NHEADS + h)) * SEQ + t) * HDIM + d] =
            (__bf16)((acc[mt][nt][r] + bv) * sc);
      }
  }
}

// ---------------------------------------------------------------------------
// Kernel 2: causal flash attention per (128 query rows) x (batch*head).
// 8 waves own one 16-row Q tile each; 64-key blocks double-buffered in LDS.
// K staged via gfx1250 async global->LDS, V via packed-b32 transpose; both
// pipelined one block ahead of the 16 WMMAs per key block.
// ---------------------------------------------------------------------------
__global__ __launch_bounds__(256)
void attn_kernel(const __bf16* __restrict__ Q, const __bf16* __restrict__ K,
                 const __bf16* __restrict__ V, __bf16* __restrict__ AO)
{
  constexpr int LT = 72;                  // 64 + 8 pad
  __shared__ __bf16 lK[2][64 * LT];       // [key][d], double-buffered
  __shared__ __bf16 lV[2][64 * LT];       // [d][key] (transposed), double-buffered
  __shared__ __bf16 lP[8 * 16 * LT];      // per-wave P scratch [row][key]

  const int tid  = threadIdx.x;
  const int lane = tid & 31;
  const int wave = tid >> 5;
  const int lo16 = lane & 15;
  const int hi   = lane >> 4;

  const int bh = blockIdx.y;       // 0..63
  const int b  = bh >> 4;
  const int h  = bh & 15;
  const int q0 = blockIdx.x * 128;
  const int qr = q0 + wave * 16;   // this wave's first query row

  const __bf16* Qh = Q + (size_t)bh * SEQ * HDIM;
  const __bf16* Kh = K + (size_t)bh * SEQ * HDIM;
  const __bf16* Vh = V + (size_t)bh * SEQ * HDIM;

  // Q A-fragments held in registers (head dim 64 = two K=32 chunks)
  bf16x16 aq[2];
  {
    const __bf16* qp = Qh + (size_t)(qr + lo16) * HDIM;
#pragma unroll
    for (int ks = 0; ks < 2; ++ks) {
      bf16x8 g0 = *(const bf16x8*)(qp + ks * 32 + 8 * hi);
      bf16x8 g1 = *(const bf16x8*)(qp + ks * 32 + 16 + 8 * hi);
      aq[ks] = join8(g0, g1);
    }
  }

  f32x8 o[4];
#pragma unroll
  for (int j = 0; j < 4; ++j) o[j] = {};
  float mrow[8], lrow[8];
#pragma unroll
  for (int r = 0; r < 8; ++r) { mrow[r] = -1e30f; lrow[r] = 0.0f; }

  const int nkb  = (q0 + 128) >> 6;   // causal: 64-key blocks up to block max row
  const int skey = tid >> 2;          // async K staging: key 0..63
  const int sdk  = (tid & 3) * 16;    // d base 0/16/32/48
  const int vkp  = tid >> 3;          // V staging: key-pair 0..31
  const int sdv  = (tid & 7) * 8;     // d base 0..56
  __bf16* lPw = &lP[wave * 16 * LT];

  // prologue: start block 0 transfers
  {
    const __bf16* ksrc = Kh + (size_t)skey * HDIM + sdk;
    async_copy_b128(ksrc,     &lK[0][skey * LT + sdk]);
    async_copy_b128(ksrc + 8, &lK[0][skey * LT + sdk + 8]);
  }
  bf16x8 pv0 = *(const bf16x8*)(Vh + (size_t)(2 * vkp) * HDIM + sdv);
  bf16x8 pv1 = *(const bf16x8*)(Vh + (size_t)(2 * vkp + 1) * HDIM + sdv);

  for (int kb = 0; kb < nkb; ++kb) {
    const int kpos = kb * 64;
    const int cur  = kb & 1;
    const __bf16* lKc = lK[cur];
    const __bf16* lVc = lV[cur];

    // commit prefetched V block (packed key-pair b32 stores)
#pragma unroll
    for (int i = 0; i < 8; ++i) {
      bf16x2 t; t[0] = pv0[i]; t[1] = pv1[i];
      *(bf16x2*)&lV[cur][(sdv + i) * LT + 2 * vkp] = t;
    }
    wait_async0();          // K block for `cur` now resident
    __syncthreads();

    // start next block's transfers while we compute on `cur`
    if (kb + 1 < nkb) {
      const __bf16* ksrc = Kh + (size_t)(kpos + 64 + skey) * HDIM + sdk;
      async_copy_b128(ksrc,     &lK[1 - cur][skey * LT + sdk]);
      async_copy_b128(ksrc + 8, &lK[1 - cur][skey * LT + sdk + 8]);
      pv0 = *(const bf16x8*)(Vh + (size_t)(kpos + 64 + 2 * vkp) * HDIM + sdv);
      pv1 = *(const bf16x8*)(Vh + (size_t)(kpos + 64 + 2 * vkp + 1) * HDIM + sdv);
    }

    // S = Q * K^T  (four 16-key tiles x two k-steps = 8 WMMA)
    f32x8 s[4];
#pragma unroll
    for (int nt = 0; nt < 4; ++nt) s[nt] = {};
#pragma unroll
    for (int ks = 0; ks < 2; ++ks)
#pragma unroll
      for (int nt = 0; nt < 4; ++nt) {
        const int key = nt * 16 + lo16;
        bf16x8 g0 = *(const bf16x8*)&lKc[key * LT + ks * 32 + 8 * hi];
        bf16x8 g1 = *(const bf16x8*)&lKc[key * LT + ks * 32 + 16 + 8 * hi];
        s[nt] = wmma_bf16(aq[ks], join8(g0, g1), s[nt]);
      }

    // causal mask + online softmax (row spread over 16 lanes of a half-wave)
    float p[4][8], scl[8];
#pragma unroll
    for (int r = 0; r < 8; ++r) {
      const int row = qr + r + 8 * hi;
      float v[4];
#pragma unroll
      for (int nt = 0; nt < 4; ++nt) {
        const int key = kpos + nt * 16 + lo16;
        v[nt] = (key <= row) ? s[nt][r] : -1e30f;
      }
      float t = fmaxf(fmaxf(v[0], v[1]), fmaxf(v[2], v[3]));
#pragma unroll
      for (int mm = 1; mm < 16; mm <<= 1) t = fmaxf(t, __shfl_xor(t, mm, 32));
      const float mnew = fmaxf(mrow[r], t);
      scl[r]  = __expf(mrow[r] - mnew);
      mrow[r] = mnew;
      float rs = 0.0f;
#pragma unroll
      for (int nt = 0; nt < 4; ++nt) { p[nt][r] = __expf(v[nt] - mnew); rs += p[nt][r]; }
#pragma unroll
      for (int mm = 1; mm < 16; mm <<= 1) rs += __shfl_xor(rs, mm, 32);
      lrow[r] = lrow[r] * scl[r] + rs;
    }
#pragma unroll
    for (int j = 0; j < 4; ++j)
#pragma unroll
      for (int r = 0; r < 8; ++r) o[j][r] *= scl[r];

    // reshape P: C-layout regs -> per-wave LDS -> A-layout fragments
#pragma unroll
    for (int r = 0; r < 8; ++r) {
      const int row = r + 8 * hi;
#pragma unroll
      for (int nt = 0; nt < 4; ++nt)
        lPw[row * LT + nt * 16 + lo16] = (__bf16)p[nt][r];
    }
    asm volatile("s_wait_dscnt 0" ::: "memory");  // same-wave LDS RAW fence
    bf16x16 ap[2];
#pragma unroll
    for (int c = 0; c < 2; ++c) {
      bf16x8 g0 = *(const bf16x8*)&lPw[lo16 * LT + c * 32 + 8 * hi];
      bf16x8 g1 = *(const bf16x8*)&lPw[lo16 * LT + c * 32 + 16 + 8 * hi];
      ap[c] = join8(g0, g1);
    }

    // O += P @ V  (two k-steps x four output-dim tiles = 8 WMMA)
#pragma unroll
    for (int c = 0; c < 2; ++c)
#pragma unroll
      for (int j = 0; j < 4; ++j) {
        const int d = j * 16 + lo16;
        bf16x8 g0 = *(const bf16x8*)&lVc[d * LT + c * 32 + 8 * hi];
        bf16x8 g1 = *(const bf16x8*)&lVc[d * LT + c * 32 + 16 + 8 * hi];
        o[j] = wmma_bf16(ap[c], join8(g0, g1), o[j]);
      }
    __syncthreads();
  }

  // normalize, store as bf16 into AO [8192, 1024] (row = b*T+t, col = h*64+d)
#pragma unroll
  for (int r = 0; r < 8; ++r) {
    const int row = qr + r + 8 * hi;
    const float inv = 1.0f / lrow[r];
#pragma unroll
    for (int j = 0; j < 4; ++j) {
      const int col = h * HDIM + j * 16 + lo16;
      AO[((size_t)b * SEQ + row) * DMODEL + col] = (__bf16)(o[j][r] * inv);
    }
  }
}

// ---------------------------------------------------------------------------
// Kernel 3: output projection. out[8192,1024] = AO(bf16) @ Wout + bias (f32).
// Same pipelined 128x128 / K=64 tiling as kernel 1; A is already bf16.
// ---------------------------------------------------------------------------
__global__ __launch_bounds__(256)
void out_gemm_kernel(const __bf16* __restrict__ A, const float* __restrict__ w,
                     const float* __restrict__ bias, float* __restrict__ out)
{
  constexpr int KD  = DMODEL;
  constexpr int N   = DMODEL;
  constexpr int LDT = 72;
  __shared__ __bf16 lA[128 * LDT];
  __shared__ __bf16 lB[128 * LDT];

  const int tid  = threadIdx.x;
  const int lane = tid & 31;
  const int wave = tid >> 5;
  const int wm   = wave >> 1;
  const int wn   = wave & 1;
  const int lo16 = lane & 15;
  const int hi   = lane >> 4;

  const int m0 = blockIdx.y * 128;
  const int n0 = blockIdx.x * 128;

  f32x8 acc[2][4];
#pragma unroll
  for (int i = 0; i < 2; ++i)
#pragma unroll
    for (int j = 0; j < 4; ++j) acc[i][j] = {};

  const int arow = tid >> 1;
  const int acol = (tid & 1) * 32;
  const int wkp  = tid >> 3;          // k-pair 0..31
  const int ncol = (tid & 7) * 16;    // 0..112

  bf16x8 pa[4];
  f32x4  pw0[4], pw1[4];
  {
    const __bf16* asrc = A + (size_t)(m0 + arow) * KD + acol;
    const float*  wsrc = w + (size_t)(2 * wkp) * N + n0 + ncol;
#pragma unroll
    for (int j = 0; j < 4; ++j) pa[j] = *(const bf16x8*)(asrc + 8 * j);
#pragma unroll
    for (int i = 0; i < 4; ++i) pw0[i] = *(const f32x4*)(wsrc + 4 * i);
#pragma unroll
    for (int i = 0; i < 4; ++i) pw1[i] = *(const f32x4*)(wsrc + N + 4 * i);
  }

  for (int kk = 0; kk < KD; kk += 64) {
#pragma unroll
    for (int j = 0; j < 4; ++j)
      *(bf16x8*)&lA[arow * LDT + acol + 8 * j] = pa[j];
#pragma unroll
    for (int i = 0; i < 16; ++i) {
      bf16x2 t;
      t[0] = (__bf16)pw0[i >> 2][i & 3];
      t[1] = (__bf16)pw1[i >> 2][i & 3];
      *(bf16x2*)&lB[(ncol + i) * LDT + 2 * wkp] = t;
    }
    __syncthreads();

    if (kk + 64 < KD) {
      const __bf16* asrc = A + (size_t)(m0 + arow) * KD + kk + 64 + acol;
      const float*  wsrc = w + (size_t)(kk + 64 + 2 * wkp) * N + n0 + ncol;
#pragma unroll
      for (int j = 0; j < 4; ++j) pa[j] = *(const bf16x8*)(asrc + 8 * j);
#pragma unroll
      for (int i = 0; i < 4; ++i) pw0[i] = *(const f32x4*)(wsrc + 4 * i);
#pragma unroll
      for (int i = 0; i < 4; ++i) pw1[i] = *(const f32x4*)(wsrc + N + 4 * i);
    }

#pragma unroll
    for (int ks = 0; ks < 2; ++ks) {
      bf16x16 afrag[2], bfrag[4];
#pragma unroll
      for (int mt = 0; mt < 2; ++mt) {
        const int row = wm * 32 + mt * 16 + lo16;
        bf16x8 g0 = *(const bf16x8*)&lA[row * LDT + ks * 32 + 8 * hi];
        bf16x8 g1 = *(const bf16x8*)&lA[row * LDT + ks * 32 + 16 + 8 * hi];
        afrag[mt] = join8(g0, g1);
      }
#pragma unroll
      for (int nt = 0; nt < 4; ++nt) {
        const int col = wn * 64 + nt * 16 + lo16;
        bf16x8 g0 = *(const bf16x8*)&lB[col * LDT + ks * 32 + 8 * hi];
        bf16x8 g1 = *(const bf16x8*)&lB[col * LDT + ks * 32 + 16 + 8 * hi];
        bfrag[nt] = join8(g0, g1);
      }
#pragma unroll
      for (int mt = 0; mt < 2; ++mt)
#pragma unroll
        for (int nt = 0; nt < 4; ++nt)
          acc[mt][nt] = wmma_bf16(afrag[mt], bfrag[nt], acc[mt][nt]);
    }
    __syncthreads();
  }

#pragma unroll
  for (int nt = 0; nt < 4; ++nt) {
    const int n = n0 + wn * 64 + nt * 16 + lo16;
    const float bv = bias[n];
#pragma unroll
    for (int mt = 0; mt < 2; ++mt)
#pragma unroll
      for (int r = 0; r < 8; ++r) {
        const int m = m0 + wm * 32 + mt * 16 + r + 8 * hi;
        out[(size_t)m * N + n] = acc[mt][nt][r] + bv;
      }
  }
}

// ---------------------------------------------------------------------------
extern "C" void kernel_launch(void* const* d_in, const int* in_sizes, int n_in,
                              void* d_out, int out_size, void* d_ws, size_t ws_size,
                              hipStream_t stream) {
  (void)in_sizes; (void)n_in; (void)out_size; (void)ws_size;
  const float* x     = (const float*)d_in[0];
  const float* w_qkv = (const float*)d_in[1];
  const float* b_qkv = (const float*)d_in[2];
  const float* w_out = (const float*)d_in[3];
  const float* b_out = (const float*)d_in[4];
  float* out = (float*)d_out;

  const size_t tsz = (size_t)BATCH * NHEADS * SEQ * HDIM;  // 8,388,608 elems
  __bf16* Q  = (__bf16*)d_ws;      // 16 MB
  __bf16* K  = Q + tsz;            // 16 MB
  __bf16* V  = K + tsz;            // 16 MB
  __bf16* AO = V + tsz;            // 16 MB

  qkv_gemm_kernel<<<dim3(3 * DMODEL / 128, ROWS / 128), 256, 0, stream>>>(
      x, w_qkv, b_qkv, Q, K, V);
  attn_kernel<<<dim3(SEQ / 128, BATCH * NHEADS), 256, 0, stream>>>(Q, K, V, AO);
  out_gemm_kernel<<<dim3(DMODEL / 128, ROWS / 128), 256, 0, stream>>>(
      AO, w_out, b_out, out);
}
